// HeterogeneousGnn_8504035246517
// MI455X (gfx1250) — compile-verified, matrix operand order checked by reading, not run on hardware
//
#include <hip/hip_runtime.h>
#include <hip/hip_bf16.h>

typedef __attribute__((ext_vector_type(16))) _Float16 v16h;
typedef __attribute__((ext_vector_type(8)))  float    v8f;
typedef __attribute__((ext_vector_type(4)))  unsigned int u32x4;
typedef __attribute__((ext_vector_type(4)))  int      i32x4;
typedef __attribute__((ext_vector_type(8)))  int      i32x8;

#define DD 512
#define NB 64
#define NA 1024
#define NC 64
__device__ __forceinline__ float inv_sd() { return 0.04419417382415922f; } // 1/sqrt(512)

#if __has_builtin(__builtin_amdgcn_tensor_load_to_lds)
#define HAS_TDM 1
#else
#define HAS_TDM 0
#endif

#if __has_builtin(__builtin_amdgcn_s_wait_tensorcnt)
#define WAIT_TENSORCNT(n) __builtin_amdgcn_s_wait_tensorcnt(n)
#else
#define WAIT_TENSORCNT(n)
#endif

// ---------------- TDM: load a [rows x kwidth] f16 tile (row stride = srcStride elems)
// into LDS at lds_addr. Descriptor packing per CDNA5 ISA 8.3/8.4 (count=1, data_size=2B,
// type=2, tile_dim0=kwidth, tile_dim1=rows, tensor_dim0_stride=srcStride).
__device__ __forceinline__ void tdm_load_tile(unsigned int lds_addr, const _Float16* gsrc,
                                              int rows, int kwidth, int srcStride) {
#if HAS_TDM
  unsigned long long ga = (unsigned long long)(uintptr_t)gsrc;
  u32x4 g0;
  g0[0] = 1u;                                   // count=1, user mode
  g0[1] = lds_addr;                             // lds_addr [63:32]
  g0[2] = (unsigned int)ga;                     // global_addr [95:64]
  g0[3] = (unsigned int)((ga >> 32) & 0x01FFFFFFu) | (2u << 30); // addr hi + type=2
  unsigned int dim0 = (unsigned int)srcStride;  // tensor_dim0
  unsigned int dim1 = (unsigned int)rows;       // tensor_dim1
  unsigned long long st0 = (unsigned long long)srcStride;
  i32x8 g1;
  g1[0] = (int)(1u << 16);                      // workgroup_mask=0, data_size=1 (2 bytes)
  g1[1] = (int)((dim0 & 0xFFFFu) << 16);        // tensor_dim0[15:0] -> [63:48]
  g1[2] = (int)(((dim0 >> 16) & 0xFFFFu) | ((dim1 & 0xFFFFu) << 16));
  g1[3] = (int)(((dim1 >> 16) & 0xFFFFu) | (((unsigned int)kwidth & 0xFFFFu) << 16)); // tile_dim0
  g1[4] = (int)((unsigned int)rows & 0xFFFFu);  // tile_dim1, tile_dim2=0
  g1[5] = (int)(unsigned int)(st0 & 0xFFFFFFFFu);        // tensor_dim0_stride low
  g1[6] = (int)(unsigned int)((st0 >> 32) & 0xFFFFu);    // stride hi, dim1_stride=0
  g1[7] = 0;
  i32x4 z4 = {0, 0, 0, 0};
#if defined(__clang_major__) && (__clang_major__ >= 23)
  i32x8 z8 = {0, 0, 0, 0, 0, 0, 0, 0};
  __builtin_amdgcn_tensor_load_to_lds(g0, g1, z4, z4, z8, 0);
#else
  __builtin_amdgcn_tensor_load_to_lds(g0, g1, z4, z4, 0);
#endif
#else
  (void)lds_addr; (void)gsrc; (void)rows; (void)kwidth; (void)srcStride;
#endif
}

// ---------------- LayerNorm (row length 512) with optional f16 copy ---------------
__global__ void ln_rows_k(const float* __restrict__ x, float* __restrict__ y,
                          _Float16* __restrict__ y16,
                          const float* __restrict__ w, const float* __restrict__ b) {
  long row = blockIdx.x;
  const float* xr = x + row * DD;
  float* yr = y + row * DD;
  __shared__ float red[256];
  int t = threadIdx.x;
  float v0 = xr[t], v1 = xr[t + 256];
  red[t] = v0 + v1; __syncthreads();
  for (int s = 128; s > 0; s >>= 1) { if (t < s) red[t] += red[t + s]; __syncthreads(); }
  float mu = red[0] * (1.0f / DD);
  __syncthreads();
  float d0 = v0 - mu, d1 = v1 - mu;
  red[t] = d0 * d0 + d1 * d1; __syncthreads();
  for (int s = 128; s > 0; s >>= 1) { if (t < s) red[t] += red[t + s]; __syncthreads(); }
  float rstd = rsqrtf(red[0] * (1.0f / DD) + 1e-5f);
  float o0 = d0 * rstd * w[t] + b[t];
  float o1 = d1 * rstd * w[t + 256] + b[t + 256];
  yr[t] = o0; yr[t + 256] = o1;
  if (y16) {
    _Float16* q = y16 + row * DD;
    q[t] = (_Float16)o0; q[t + 256] = (_Float16)o1;
  }
}

// ---------------- weight convert+transpose: Wt16[n,k] = (f16) W[k,n], 512x512 ------
__global__ void convT_k(const float* __restrict__ W, _Float16* __restrict__ Wt) {
  int idx = blockIdx.x * 256 + threadIdx.x;   // over 512*512
  int n = idx >> 9, k = idx & 511;
  Wt[(long)n * DD + k] = (_Float16)W[(long)k * DD + n];
}

// ---------------- hl[b,c,d] = mean_j xln[b, c+64j, d] -----------------------------
__global__ void avg_pool_k(const float* __restrict__ xln, float* __restrict__ hl) {
  long idx = (long)blockIdx.x * blockDim.x + threadIdx.x;   // over B*C*D
  int d = idx & (DD - 1);
  int c = (int)((idx >> 9) & (NC - 1));
  long b = idx >> 15;
  const float* base = xln + ((b * NA + c) << 9) + d;
  float s = 0.0f;
#pragma unroll
  for (int j = 0; j < 16; ++j) s += base[(long)j * NC * DD];
  hl[idx] = s * (1.0f / 16.0f);
}

// ======== big projection GEMM: C = alpha * A16[M,K] x B16[N,K]^T,  M % 128 == 0 ====
// 256 thr = 8 waves (4x2) -> 128x64 block tile, 32x32 per wave (4 accumulators).
// Double-buffered LDS tiles filled by the Tensor Data Mover: the next K-step's DMA is
// issued BEFORE computing the current one, and wave0 waits tensorcnt<=2 so the two
// prefetches stream behind the WMMAs. Optional f16 copy of C.
__global__ void gemm_f16_big_k(const _Float16* __restrict__ A16,
                               const _Float16* __restrict__ B16,
                               float* __restrict__ C, _Float16* __restrict__ C16,
                               int M, int N, int K, float alpha) {
  __shared__ _Float16 tA[2][128 * 64];
  __shared__ _Float16 tB[2][64 * 64];
  int tid = threadIdx.x;
  int wave = tid >> 5, lane = tid & 31;
  int wm = wave >> 1, wn = wave & 1;
  int half = lane >> 4, r = lane & 15;
  long tm = (long)blockIdx.y * 128;
  long tn = (long)blockIdx.x * 64;
  v8f acc[2][2] = {};

#if HAS_TDM
  if (tid < 32) {                         // prologue prefetch into buffer 0
    tdm_load_tile((unsigned int)(uintptr_t)&tA[0][0], A16 + tm * K, 128, 64, K);
    tdm_load_tile((unsigned int)(uintptr_t)&tB[0][0], B16 + tn * K, 64, 64, K);
  }
#endif
  int buf = 0;
  for (int k0 = 0; k0 < K; k0 += 64, buf ^= 1) {
#if HAS_TDM
    if (tid < 32) {
      if (k0 + 64 < K) {                  // prefetch next K-step into other buffer
        tdm_load_tile((unsigned int)(uintptr_t)&tA[buf ^ 1][0], A16 + tm * K + k0 + 64, 128, 64, K);
        tdm_load_tile((unsigned int)(uintptr_t)&tB[buf ^ 1][0], B16 + tn * K + k0 + 64, 64, 64, K);
        WAIT_TENSORCNT(2);                // current tile done; 2 prefetches in flight
      } else {
        WAIT_TENSORCNT(0);
      }
    }
    __syncthreads();
#else
    for (int idx = tid; idx < 128 * 8; idx += 256) {
      int row = idx >> 3, ch = (idx & 7) << 3;
      *(float4*)&tA[buf][row * 64 + ch] = *(const float4*)&A16[(tm + row) * K + k0 + ch];
      if (row < 64)
        *(float4*)&tB[buf][row * 64 + ch] = *(const float4*)&B16[(tn + row) * K + k0 + ch];
    }
    __syncthreads();
#endif
#pragma unroll
    for (int kk = 0; kk < 64; kk += 32) {
      v16h a[2], b[2];
#pragma unroll
      for (int rt = 0; rt < 2; ++rt) {
        const _Float16* ar = &tA[buf][(wm * 32 + rt * 16 + r) * 64 + kk];
        const _Float16* br = &tB[buf][(wn * 32 + rt * 16 + r) * 64 + kk];
#pragma unroll
        for (int i = 0; i < 8; ++i) {
          a[rt][i]     = ar[half * 8 + i];
          a[rt][8 + i] = ar[16 + half * 8 + i];
          b[rt][i]     = br[half * 8 + i];
          b[rt][8 + i] = br[16 + half * 8 + i];
        }
      }
#pragma unroll
      for (int rt = 0; rt < 2; ++rt)
#pragma unroll
        for (int ct = 0; ct < 2; ++ct)
          acc[rt][ct] = __builtin_amdgcn_wmma_f32_16x16x32_f16(
              false, a[rt], false, b[ct], (short)0, acc[rt][ct], false, false);
    }
    __syncthreads();                      // tile consumed before next DMA overwrites
  }
#pragma unroll
  for (int rt = 0; rt < 2; ++rt) {
#pragma unroll
    for (int ct = 0; ct < 2; ++ct) {
      long m0 = tm + wm * 32 + rt * 16 + half * 8;
      long n0 = tn + wn * 32 + ct * 16 + r;
      float* cp = C + m0 * N + n0;
#pragma unroll
      for (int i = 0; i < 8; ++i) cp[(long)i * N] = acc[rt][ct][i] * alpha;
      if (C16) {
        _Float16* cq = C16 + m0 * N + n0;
#pragma unroll
        for (int i = 0; i < 8; ++i) cq[(long)i * N] = (_Float16)(acc[rt][ct][i] * alpha);
      }
    }
  }
}

// ---------------- batched 64x64 f16 GEMM (used for s = hide x hide^T) --------------
__global__ void gemm_f16_fast_k(const _Float16* __restrict__ A16,
                                const _Float16* __restrict__ B16,
                                float* __restrict__ C, _Float16* __restrict__ C16,
                                int M, int N, int K,
                                long sA, long sB, long sC, float alpha) {
  __shared__ _Float16 tA[64 * 64];
  __shared__ _Float16 tB[64 * 64];
  int tid = threadIdx.x;
  int wave = tid >> 5, lane = tid & 31;
  int wm = wave >> 1, wn = wave & 1;
  int half = lane >> 4, r = lane & 15;
  long tm = (long)blockIdx.y * 64;
  long tn = (long)blockIdx.x * 64;
  const _Float16* Ab = A16 + (long)blockIdx.z * sA;
  const _Float16* Bb = B16 + (long)blockIdx.z * sB;
  v8f acc0 = {}, acc1 = {};

  for (int k0 = 0; k0 < K; k0 += 64) {
#if HAS_TDM
    if (tid < 32) {
      tdm_load_tile((unsigned int)(uintptr_t)tA, Ab + tm * K + k0, 64, 64, K);
      tdm_load_tile((unsigned int)(uintptr_t)tB, Bb + tn * K + k0, 64, 64, K);
      WAIT_TENSORCNT(0);
    }
    __syncthreads();
#else
    for (int idx = tid; idx < 64 * 8; idx += 256) {
      int row = idx >> 3, ch = (idx & 7) << 3;
      *(float4*)&tA[row * 64 + ch] = *(const float4*)&Ab[(tm + row) * K + k0 + ch];
      *(float4*)&tB[row * 64 + ch] = *(const float4*)&Bb[(tn + row) * K + k0 + ch];
    }
    __syncthreads();
#endif
#pragma unroll
    for (int kk = 0; kk < 64; kk += 32) {
      v16h a, b0, b1;
      const _Float16* ar  = &tA[(wm * 16 + r) * 64 + kk];
      const _Float16* br0 = &tB[(wn * 32 + r) * 64 + kk];
      const _Float16* br1 = &tB[(wn * 32 + 16 + r) * 64 + kk];
#pragma unroll
      for (int i = 0; i < 8; ++i) {
        a[i]      = ar[half * 8 + i];
        a[8 + i]  = ar[16 + half * 8 + i];
        b0[i]     = br0[half * 8 + i];
        b0[8 + i] = br0[16 + half * 8 + i];
        b1[i]     = br1[half * 8 + i];
        b1[8 + i] = br1[16 + half * 8 + i];
      }
      acc0 = __builtin_amdgcn_wmma_f32_16x16x32_f16(false, a, false, b0, (short)0, acc0, false, false);
      acc1 = __builtin_amdgcn_wmma_f32_16x16x32_f16(false, a, false, b1, (short)0, acc1, false, false);
    }
    __syncthreads();
  }
  long m0 = tm + wm * 16 + half * 8;
  long n0 = tn + wn * 32 + r;
  float* cp = C + (long)blockIdx.z * sC + m0 * N + n0;
#pragma unroll
  for (int i = 0; i < 8; ++i) {
    cp[(long)i * N] = acc0[i] * alpha;
    cp[(long)i * N + 16] = acc1[i] * alpha;
  }
  if (C16) {
    _Float16* cq = C16 + (long)blockIdx.z * sC + m0 * N + n0;
#pragma unroll
    for (int i = 0; i < 8; ++i) {
      cq[(long)i * N] = (_Float16)(acc0[i] * alpha);
      cq[(long)i * N + 16] = (_Float16)(acc1[i] * alpha);
    }
  }
}

// ---------------- generic slow WMMA GEMM (f32 in), used only for s @ hln (K=64) ---
template <bool TB>
__global__ void gemm_wmma_k(const float* __restrict__ A, const float* __restrict__ B,
                            float* __restrict__ C, int M, int N, int K,
                            long sA, long sB, long sC, float alpha) {
  int wave = threadIdx.x >> 5;
  int lane = threadIdx.x & 31;
  int tn = (blockIdx.x * 8 + wave) * 16;
  int tm = blockIdx.y * 16;
  if (tn >= N || tm >= M) return;     // wave-uniform exit keeps EXEC all-ones for WMMA
  const float* Ab = A + (long)blockIdx.z * sA;
  const float* Bb = B + (long)blockIdx.z * sB;
  float*       Cb = C + (long)blockIdx.z * sC;
  int half = lane >> 4;
  int r    = lane & 15;
  v8f acc = {};
  for (int k0 = 0; k0 < K; k0 += 32) {
    v16h a, bfrag;
    const float* arow = Ab + (long)(tm + r) * K + k0;
#pragma unroll
    for (int i = 0; i < 8; ++i) {
      a[i]     = (_Float16)arow[half * 8 + i];
      a[8 + i] = (_Float16)arow[16 + half * 8 + i];
    }
    if (TB) {
      const float* brow = Bb + (long)(tn + r) * K + k0;
#pragma unroll
      for (int i = 0; i < 8; ++i) {
        bfrag[i]     = (_Float16)brow[half * 8 + i];
        bfrag[8 + i] = (_Float16)brow[16 + half * 8 + i];
      }
    } else {
      const float* bcol = Bb + (long)k0 * N + (tn + r);
#pragma unroll
      for (int i = 0; i < 8; ++i) {
        bfrag[i]     = (_Float16)bcol[(long)(half * 8 + i) * N];
        bfrag[8 + i] = (_Float16)bcol[(long)(16 + half * 8 + i) * N];
      }
    }
    acc = __builtin_amdgcn_wmma_f32_16x16x32_f16(false, a, false, bfrag,
                                                 (short)0, acc, false, false);
  }
  float* crow = Cb + (long)(tm + half * 8) * N + tn + r;
#pragma unroll
  for (int i = 0; i < 8; ++i) crow[(long)i * N] = acc[i] * alpha;
}

// ---------------- row softmax, width 64 -------------------------------------------
__global__ void softmax64_k(float* __restrict__ s) {
  long row = blockIdx.x;
  float* p = s + row * 64;
  __shared__ float red[64];
  int t = threadIdx.x;
  float v = p[t];
  red[t] = v; __syncthreads();
  for (int q = 32; q > 0; q >>= 1) { if (t < q) red[t] = fmaxf(red[t], red[t + q]); __syncthreads(); }
  float m = red[0]; __syncthreads();
  float e = __expf(v - m);
  red[t] = e; __syncthreads();
  for (int q = 32; q > 0; q >>= 1) { if (t < q) red[t] += red[t + q]; __syncthreads(); }
  p[t] = e / red[0];
}

// ------- h2a with 2-entry masked softmax: out[b,a] = w0*hw[b,a%64] + w1*att[b,a] --
__global__ void h2a_combine_k(const float* __restrict__ att, const float* __restrict__ hw,
                              float* __restrict__ out) {
  int a = blockIdx.x & (NA - 1);
  long b = blockIdx.x >> 10;
  int c = a & (NC - 1);
  const float* ar = att + ((b * NA + a) << 9);
  const float* hr = hw  + ((b * NC + c) << 9);
  float* orow     = out + ((b * NA + a) << 9);
  __shared__ float r0[128], r1[128];
  int t = threadIdx.x;
  float p0 = 0.0f, p1 = 0.0f;
  for (int i = t; i < DD; i += 128) {
    float av = ar[i];
    p0 += av * hr[i];
    p1 += av * av;
  }
  r0[t] = p0; r1[t] = p1; __syncthreads();
  for (int q = 64; q > 0; q >>= 1) {
    if (t < q) { r0[t] += r0[t + q]; r1[t] += r1[t + q]; }
    __syncthreads();
  }
  float s0 = r0[0] * inv_sd(), s1 = r1[0] * inv_sd();
  float m  = fmaxf(s0, s1);
  float e0 = __expf(s0 - m), e1 = __expf(s1 - m);
  float inv = 1.0f / (e0 + e1);
  float w0 = e0 * inv, w1 = e1 * inv;
  for (int i = t; i < DD; i += 128) orow[i] = w0 * hr[i] + w1 * ar[i];
}

// ------- layer-2 a2h attention with 17-entry masked softmax ----------------------
__global__ void a2h_attn_k(const float* __restrict__ hl1, const float* __restrict__ g,
                           float* __restrict__ out) {
  int c = blockIdx.x & (NC - 1);
  long b = blockIdx.x >> 6;
  const float* hr = hl1 + ((b * NC + c) << 9);
  const float* gb = g + ((b * NA) << 9);
  float* orow = out + ((b * NC + c) << 9);
  __shared__ float red[256];
  __shared__ float dots[17];
  int t = threadIdx.x;
  float part[17];
#pragma unroll
  for (int j = 0; j < 17; ++j) part[j] = 0.0f;
  for (int i = t; i < DD; i += 256) {
    float hv = hr[i];
#pragma unroll
    for (int j = 0; j < 16; ++j) part[j] += hv * gb[(((long)(c + 64 * j)) << 9) + i];
    part[16] += hv * hv;
  }
  for (int j = 0; j < 17; ++j) {
    red[t] = part[j]; __syncthreads();
    for (int q = 128; q > 0; q >>= 1) { if (t < q) red[t] += red[t + q]; __syncthreads(); }
    if (t == 0) dots[j] = red[0];
    __syncthreads();
  }
  float m = -1e30f, sc[17];
#pragma unroll
  for (int j = 0; j < 17; ++j) { sc[j] = dots[j] * inv_sd(); m = fmaxf(m, sc[j]); }
  float sum = 0.0f;
#pragma unroll
  for (int j = 0; j < 17; ++j) { sc[j] = __expf(sc[j] - m); sum += sc[j]; }
  float inv = 1.0f / sum;
  for (int i = t; i < DD; i += 256) {
    float acc = sc[16] * inv * hr[i];
#pragma unroll
    for (int j = 0; j < 16; ++j) acc += sc[j] * inv * gb[(((long)(c + 64 * j)) << 9) + i];
    orow[i] = acc;
  }
}

extern "C" void kernel_launch(void* const* d_in, const int* in_sizes, int n_in,
                              void* d_out, int out_size, void* d_ws, size_t ws_size,
                              hipStream_t stream) {
  (void)in_sizes; (void)n_in; (void)out_size; (void)ws_size;
  const float* att_vf   = (const float*)d_in[0];
  const float* l1_W_h2h = (const float*)d_in[4];
  const float* l1_W_h2a = (const float*)d_in[5];
  const float* l1_ln1_w = (const float*)d_in[6];  const float* l1_ln1_b = (const float*)d_in[7];
  const float* l1_ln2_w = (const float*)d_in[8];  const float* l1_ln2_b = (const float*)d_in[9];
  const float* l1_ln3_w = (const float*)d_in[10]; const float* l1_ln3_b = (const float*)d_in[11];
  const float* l2_W_a2h = (const float*)d_in[12];
  const float* l2_W_h2h = (const float*)d_in[13];
  const float* l2_W_h2a = (const float*)d_in[14];
  const float* l2_ln1_w = (const float*)d_in[15]; const float* l2_ln1_b = (const float*)d_in[16];
  const float* l2_ln2_w = (const float*)d_in[17]; const float* l2_ln2_b = (const float*)d_in[18];
  const float* l2_ln3_w = (const float*)d_in[19]; const float* l2_ln3_b = (const float*)d_in[20];
  float* out = (float*)d_out;

  const long BIG = (long)NB * NA * DD;   // 33,554,432
  const long SM  = (long)NB * NC * DD;   //  2,097,152
  const long WW  = (long)DD * DD;        //    262,144
  float* W    = (float*)d_ws;
  float* xln  = W;                 // [B,A,D] (x_ln, later att_ln)
  float* att1 = W + BIG;           // [B,A,D] att1
  float* hl   = att1 + BIG;        // [B,C,D]
  float* hln  = hl + SM;           // [B,C,D]
  float* hide = hln + SM;          // [B,C,D]
  float* hl1  = hide + SM;         // [B,C,D]
  float* hl2  = hl1 + SM;          // [B,C,D]
  float* sbuf = hl2 + SM;          // [B,C,C]
  _Float16* f16base = (_Float16*)(sbuf + (long)NB * NC * NC);
  _Float16* xln16  = f16base;            // [B,A,D] f16
  _Float16* hln16  = xln16 + BIG;        // [B,C,D] f16
  _Float16* hide16 = hln16 + SM;         // [B,C,D] f16
  _Float16* wt_h2h1 = hide16 + SM;       // 5 transposed f16 weights [N,K]
  _Float16* wt_h2a1 = wt_h2h1 + WW;
  _Float16* wt_a2h2 = wt_h2a1 + WW;
  _Float16* wt_h2h2 = wt_a2h2 + WW;
  _Float16* wt_h2a2 = wt_h2h2 + WW;

  const float isd = 0.04419417382415922f;
  const long sHD = (long)NC * DD, sSS = (long)NC * NC;

  // weight conversion (f32 [K,N] -> f16 [N,K])
  convT_k<<<1024, 256, 0, stream>>>(l1_W_h2h, wt_h2h1);
  convT_k<<<1024, 256, 0, stream>>>(l1_W_h2a, wt_h2a1);
  convT_k<<<1024, 256, 0, stream>>>(l2_W_a2h, wt_a2h2);
  convT_k<<<1024, 256, 0, stream>>>(l2_W_h2h, wt_h2h2);
  convT_k<<<1024, 256, 0, stream>>>(l2_W_h2a, wt_h2a2);

  // ---------------- layer 1 ----------------
  ln_rows_k<<<NB * NA, 256, 0, stream>>>(att_vf, xln, nullptr, l1_ln1_w, l1_ln1_b);
  avg_pool_k<<<(NB * NC * DD) / 256, 256, 0, stream>>>(xln, hl);
  // h2h
  ln_rows_k<<<NB * NC, 256, 0, stream>>>(hl, hln, hln16, l1_ln2_w, l1_ln2_b);
  gemm_f16_big_k<<<dim3(8, 32, 1), 256, 0, stream>>>(
      hln16, wt_h2h1, hide, hide16, NB * NC, DD, DD, 1.0f);
  gemm_f16_fast_k<<<dim3(1, 1, NB), 256, 0, stream>>>(
      hide16, hide16, sbuf, nullptr, NC, NC, DD, sHD, sHD, sSS, isd);
  softmax64_k<<<NB * NC, 64, 0, stream>>>(sbuf);
  gemm_wmma_k<false><<<dim3(4, NC / 16, NB), 256, 0, stream>>>(
      sbuf, hln, hl1, NC, DD, NC, sSS, sHD, sHD, 1.0f);
  // h2a (2-entry masked softmax)
  ln_rows_k<<<NB * NC, 256, 0, stream>>>(hl1, hln, hln16, l1_ln3_w, l1_ln3_b);
  gemm_f16_big_k<<<dim3(8, 32, 1), 256, 0, stream>>>(
      hln16, wt_h2a1, hide, nullptr, NB * NC, DD, DD, 1.0f);
  h2a_combine_k<<<NB * NA, 128, 0, stream>>>(att_vf, hide, att1);

  // ---------------- layer 2 ----------------
  ln_rows_k<<<NB * NA, 256, 0, stream>>>(att1, xln, xln16, l2_ln1_w, l2_ln1_b);
  gemm_f16_big_k<<<dim3(8, 512, 1), 256, 0, stream>>>(
      xln16, wt_a2h2, out, nullptr, NB * NA, DD, DD, 1.0f);   // g in d_out
  a2h_attn_k<<<NB * NC, 256, 0, stream>>>(hl1, out, hl2);
  // h2h
  ln_rows_k<<<NB * NC, 256, 0, stream>>>(hl2, hln, hln16, l2_ln2_w, l2_ln2_b);
  gemm_f16_big_k<<<dim3(8, 32, 1), 256, 0, stream>>>(
      hln16, wt_h2h2, hide, hide16, NB * NC, DD, DD, 1.0f);
  gemm_f16_fast_k<<<dim3(1, 1, NB), 256, 0, stream>>>(
      hide16, hide16, sbuf, nullptr, NC, NC, DD, sHD, sHD, sSS, isd);
  softmax64_k<<<NB * NC, 64, 0, stream>>>(sbuf);
  gemm_wmma_k<false><<<dim3(4, NC / 16, NB), 256, 0, stream>>>(
      sbuf, hln, hl, NC, DD, NC, sSS, sHD, sHD, 1.0f);
  // h2a -> final output
  ln_rows_k<<<NB * NC, 256, 0, stream>>>(hl, hln, hln16, l2_ln3_w, l2_ln3_b);
  gemm_f16_big_k<<<dim3(8, 32, 1), 256, 0, stream>>>(
      hln16, wt_h2a2, hide, nullptr, NB * NC, DD, DD, 1.0f);
  h2a_combine_k<<<NB * NA, 128, 0, stream>>>(att1, hide, out);
}
